// HighFidelityKANLayer_26929444946284
// MI455X (gfx1250) — matrix operand order repaired and checked by reading, not compiled, not run on hardware
//
#include <hip/hip_runtime.h>
#include <hip/hip_bf16.h>
#include <math.h>

// Problem sizes (fixed by the reference)
#define BSZ 16384
#define DD  512
#define OO  512
#define KT  4608         // combined K: 512 (base) + 4096 (spline)
#define NCH 72           // KT / KC
// Tiling
#define BM  128          // batch rows per block
#define BN  256          // output cols per block
#define KC  64           // K elements (bf16) staged per chunk
#define LDK 72           // KC + 8 bf16 pad -> row stride 144 B = 36 dwords (conflict-free)

typedef __attribute__((ext_vector_type(16))) __bf16 v16bf;
typedef __attribute__((ext_vector_type(8)))  __bf16 v8bf;
typedef __attribute__((ext_vector_type(8)))  float  v8f;

__device__ __forceinline__ float silu_f(float x) {
    return x * (1.0f / (1.0f + __expf(-x)));
}

// Cubic B-spline basis on the reference's uniform grid:
// knots t[j] = -3.2 + 0.4*j, j = 0..11 ; returns 8 degree-3 values.
__device__ __forceinline__ void bspline8(float x, float* __restrict__ o) {
    const float r1 = 1.0f / (0.4f + 1e-9f);
    const float r2 = 1.0f / (0.8f + 1e-9f);
    const float r3 = 1.0f / (1.2f + 1e-9f);
    float b0[11];
#pragma unroll
    for (int j = 0; j < 11; ++j) {
        const float tj = -3.2f + 0.4f * (float)j;
        b0[j] = (x >= tj && x < tj + 0.4f) ? 1.0f : 0.0f;
    }
    float b1[10];
#pragma unroll
    for (int j = 0; j < 10; ++j) {
        const float tj = -3.2f + 0.4f * (float)j;
        b1[j] = (x - tj) * r1 * b0[j] + ((tj + 0.8f) - x) * r1 * b0[j + 1];
    }
    float b2[9];
#pragma unroll
    for (int j = 0; j < 9; ++j) {
        const float tj = -3.2f + 0.4f * (float)j;
        b2[j] = (x - tj) * r2 * b1[j] + ((tj + 1.2f) - x) * r2 * b1[j + 1];
    }
#pragma unroll
    for (int j = 0; j < 8; ++j) {
        const float tj = -3.2f + 0.4f * (float)j;
        o[j] = (x - tj) * r3 * b2[j] + ((tj + 1.6f) - x) * r3 * b2[j + 1];
    }
}

// gfx1250 async DMA: 16 B global -> LDS, tracked by ASYNCcnt.
// Low 32 bits of a flat LDS address are the LDS byte offset (LDS aperture).
__device__ __forceinline__ void async_ld16(const void* gsrc, void* ldst) {
    unsigned lds = (unsigned)(uintptr_t)ldst;
    asm volatile("global_load_async_to_lds_b128 %0, %1, off"
                 :: "v"(lds), "v"(gsrc) : "memory");
}
__device__ __forceinline__ void wait_async(bool pending8) {
    if (pending8) asm volatile("s_wait_asynccnt 0x8" ::: "memory");
    else          asm volatile("s_wait_asynccnt 0x0" ::: "memory");
}

// Pre-pack weights once: W2[o][0..511] = base_weight[o][:], W2[o][512..4607] = spline flat.
__global__ __launch_bounds__(256)
void pack_w2(const float* __restrict__ wb, const float* __restrict__ ws,
             __bf16* __restrict__ w2) {
    const int idx = blockIdx.x * 256 + threadIdx.x;     // grid sized exactly OO*KT/256
    const int o = idx / KT, j = idx - o * KT;
    const float v = (j < DD) ? wb[o * DD + j]
                             : ws[(size_t)o * (DD * 8) + (j - DD)];
    w2[idx] = (__bf16)v;
}

__global__ __launch_bounds__(256)
void kan_fused_wmma(const float* __restrict__ x,
                    const float* __restrict__ wb,       // [O, D]       (fallback path)
                    const float* __restrict__ ws,       // [O, D*8]     (fallback path)
                    const __bf16* __restrict__ w2,      // [O, 4608] bf16 or nullptr
                    float* __restrict__ out) {          // [B, O]
    __shared__ __align__(16) __bf16 sA[BM][LDK];        // 18 KB
    __shared__ __align__(16) __bf16 sB[2][BN][LDK];     // 72 KB, double-buffered

    const int tid  = threadIdx.x;
    const int lane = tid & 31;
    const int wv   = tid >> 5;        // 8 waves
    const int wm   = wv >> 2;         // 0..1  -> 64-row strip
    const int wn   = wv & 3;          // 0..3  -> 64-col strip
    const int l15  = lane & 15;
    const int hsel = lane >> 4;       // half-wave select
    const int m0g  = blockIdx.y * BM;
    const int n0g  = blockIdx.x * BN;

    v8f acc[4][4];
    const v8f vzero = {0.f, 0.f, 0.f, 0.f, 0.f, 0.f, 0.f, 0.f};
#pragma unroll
    for (int i = 0; i < 4; ++i)
#pragma unroll
        for (int j = 0; j < 4; ++j) acc[i][j] = vzero;

    // Issue async DMA of one 256x64 bf16 B tile (8 x b128 per thread).
    auto issue_B = [&](int ch, int buf) {
#pragma unroll
        for (int it = 0; it < 8; ++it) {
            const int idx = tid + it * 256;
            const int row = idx >> 3, c = idx & 7;
            async_ld16(w2 + (size_t)(n0g + row) * KT + ch * KC + c * 8,
                       &sB[buf][row][c * 8]);
        }
    };

    // One KC=64 chunk of 16x16x32 bf16 WMMAs (2 k-steps x 4x4 tiles).
    auto mma_tile = [&](int buf) {
#pragma unroll
        for (int ks = 0; ks < KC / 32; ++ks) {
            v16bf af[4], bfm[4];
#pragma unroll
            for (int mt = 0; mt < 4; ++mt) {
                // A layout (ISA 7.12.2): lane m -> K{0..7,16..23}; lane m+16 -> K{8..15,24..31}
                const __bf16* p = &sA[wm * 64 + mt * 16 + l15][ks * 32];
                v8bf lo = *(const v8bf*)(p + hsel * 8);
                v8bf hi = *(const v8bf*)(p + 16 + hsel * 8);
                af[mt] = __builtin_shufflevector(lo, hi,
                         0,1,2,3,4,5,6,7,8,9,10,11,12,13,14,15);
            }
#pragma unroll
            for (int nt = 0; nt < 4; ++nt) {
                // B layout (same as A): lane n -> K{0..7,16..23}; lane n+16 -> K{8..15,24..31}
                const __bf16* p = &sB[buf][wn * 64 + nt * 16 + l15][ks * 32];
                v8bf lo = *(const v8bf*)(p + hsel * 8);
                v8bf hi = *(const v8bf*)(p + 16 + hsel * 8);
                bfm[nt] = __builtin_shufflevector(lo, hi,
                          0,1,2,3,4,5,6,7,8,9,10,11,12,13,14,15);
            }
#pragma unroll
            for (int mt = 0; mt < 4; ++mt)
#pragma unroll
                for (int nt = 0; nt < 4; ++nt)
                    acc[mt][nt] = __builtin_amdgcn_wmma_f32_16x16x32_bf16(
                        false, af[mt], false, bfm[nt],
                        (short)0, acc[mt][nt], false, false);
        }
    };

    if (w2) issue_B(0, 0);

    for (int ch = 0; ch < NCH; ++ch) {
        const int  cur  = ch & 1;
        const bool more = (ch + 1 < NCH);

        // ---- B tile: async DMA of next chunk (bf16), or fp32+convert fallback ----
        if (w2) {
            if (more) issue_B(ch + 1, cur ^ 1);
        } else {
            const float* bsrc; int bstr, bcol;
            if (ch < 8) { bsrc = wb; bstr = DD;     bcol = ch * KC; }
            else        { bsrc = ws; bstr = DD * 8; bcol = (ch - 8) * KC; }
#pragma unroll
            for (int it = 0; it < 16; ++it) {
                const int idx = tid + it * 256;
                const int row = idx >> 4, c4 = idx & 15;
                const float4 v = *(const float4*)(bsrc + (size_t)(n0g + row) * bstr + bcol + c4 * 4);
                __bf16* dst = &sB[cur][row][c4 * 4];
                dst[0] = (__bf16)v.x; dst[1] = (__bf16)v.y;
                dst[2] = (__bf16)v.z; dst[3] = (__bf16)v.w;
            }
        }

        // ---- A tile: chunks 0..7 = silu(x); chunks 8..71 = spline basis ----
        if (ch < 8) {
            const int k0g = ch * KC;
#pragma unroll
            for (int it = 0; it < 8; ++it) {
                const int idx = tid + it * 256;
                const int row = idx >> 4, c4 = idx & 15;
                const float4 v = *(const float4*)(x + (size_t)(m0g + row) * DD + k0g + c4 * 4);
                __bf16* dst = &sA[row][c4 * 4];
                dst[0] = (__bf16)silu_f(v.x);
                dst[1] = (__bf16)silu_f(v.y);
                dst[2] = (__bf16)silu_f(v.z);
                dst[3] = (__bf16)silu_f(v.w);
            }
        } else {
            const int i0 = (ch - 8) * 8;
#pragma unroll
            for (int it = 0; it < 4; ++it) {
                const int idx = tid + it * 256;
                const int row = idx >> 3, il = idx & 7;
                float xv = x[(size_t)(m0g + row) * DD + i0 + il];
                xv = fminf(fmaxf(xv, -2.0f), 2.0f);
                float bb[8];
                bspline8(xv, bb);
                __bf16* dst = &sA[row][il * 8];
#pragma unroll
                for (int c = 0; c < 8; ++c) dst[c] = (__bf16)bb[c];
            }
            if (i0 + 8 < DD)
                __builtin_prefetch(x + (size_t)(m0g + (tid >> 1)) * DD + i0 + 8, 0, 3);
        }

        if (w2) wait_async(more);     // own wave's part of B(ch) is in LDS
        __syncthreads();              // all waves' A(ch)+B(ch) visible
        mma_tile(cur);
        __syncthreads();              // safe to restage sA / reuse sB[cur] next round
    }

    // ---- Epilogue: C frag layout -> row-major f32 [B, O] ----
#pragma unroll
    for (int mt = 0; mt < 4; ++mt) {
        const int rbase = m0g + wm * 64 + mt * 16 + hsel * 8;
#pragma unroll
        for (int nt = 0; nt < 4; ++nt) {
            const int col = n0g + wn * 64 + nt * 16 + l15;
#pragma unroll
            for (int r = 0; r < 8; ++r)
                out[(size_t)(rbase + r) * OO + col] = acc[mt][nt][r];
        }
    }
}

extern "C" void kernel_launch(void* const* d_in, const int* in_sizes, int n_in,
                              void* d_out, int out_size, void* d_ws, size_t ws_size,
                              hipStream_t stream) {
    const float* x  = (const float*)d_in[0];   // [16384, 512]
    const float* wb = (const float*)d_in[1];   // [512, 512]
    const float* ws = (const float*)d_in[2];   // [512, 512, 8]
    float* out = (float*)d_out;                // [16384, 512]

    __bf16* w2 = nullptr;
    const size_t need = (size_t)OO * KT * sizeof(unsigned short);   // 4.72 MB
    if (d_ws != nullptr && ws_size >= need) {
        w2 = (__bf16*)d_ws;
        hipLaunchKernelGGL(pack_w2, dim3((OO * KT) / 256), dim3(256), 0, stream,
                           wb, ws, w2);
    }
    dim3 grid(OO / BN, BSZ / BM);              // (2, 128)
    hipLaunchKernelGGL(kan_fused_wmma, grid, dim3(256), 0, stream,
                       x, wb, ws, w2, out);
}
